// MaskedAttentionLayer_79027398246574
// MI455X (gfx1250) — matrix operand order repaired and checked by reading, not compiled, not run on hardware
//
#include <hip/hip_runtime.h>

typedef __bf16 bf16;
typedef bf16  v8bf  __attribute__((ext_vector_type(8)));
typedef bf16  v16bf __attribute__((ext_vector_type(16)));
typedef float v8f   __attribute__((ext_vector_type(8)));

#define Bn  4
#define Sn  2048
#define En  1024
#define Hn  16
#define HDn 64
#define NTOK (Bn * Sn)          // 8192 tokens
#define NEG_INF (-1e30f)

// ---------------------------------------------------------------------------
// helpers
// ---------------------------------------------------------------------------
static __device__ __forceinline__ v16bf pack16(v8bf lo, v8bf hi) {
    v16bf r;
#pragma unroll
    for (int i = 0; i < 8; ++i) { r[i] = lo[i]; r[i + 8] = hi[i]; }
    return r;
}

static __device__ __forceinline__ v8f wmma_bf16(v16bf a, v16bf b, v8f c) {
    // v_wmma_f32_16x16x32_bf16  D = A(16x32) * B(32x16) + C
    return __builtin_amdgcn_wmma_f32_16x16x32_bf16(
        /*neg_a=*/false, a, /*neg_b=*/false, b,
        /*c_mod=*/(short)0, c, /*reuse_a=*/false, /*reuse_b=*/false);
}

// async global -> LDS copy, 16 bytes per lane, tracked by ASYNCcnt
static __device__ __forceinline__ void async_b128(unsigned lds_off, const bf16* g) {
    unsigned long long ga = (unsigned long long)g;
    asm volatile("global_load_async_to_lds_b128 %0, %1, off"
                 :: "v"(lds_off), "v"(ga) : "memory");
}

// ---------------------------------------------------------------------------
// phase 1: fp32 -> bf16 conversion (x and the three weight matrices)
// ---------------------------------------------------------------------------
__global__ __launch_bounds__(256)
void cvt_f32_bf16(const float* __restrict__ src, bf16* __restrict__ dst, int n) {
    int i = blockIdx.x * blockDim.x + threadIdx.x;
    int stride = gridDim.x * blockDim.x;
    for (; i < n; i += stride) dst[i] = (bf16)src[i];
}

// ---------------------------------------------------------------------------
// phase 2: QKV projection.  out[m, c] = sum_e x[m,e] * W[c,e] + bias[c]
// Block = 256 threads (8 waves) computing a 128x128 tile. A/B staged in LDS
// via async global->LDS copies, double buffered. Each wave owns a 32x64
// register sub-tile (2 M-frags x 4 N-frags, 8 WMMA per 32-wide K step).
// Q,K stored [B,H,S,HD] bf16; V stored transposed [B,H,HD,S] bf16.
// ---------------------------------------------------------------------------
__global__ __launch_bounds__(256)
void qkv_gemm(const bf16* __restrict__ xb,          // [NTOK, E]
              const bf16* __restrict__ wb,          // [3, E(out), E(in)]
              const float* __restrict__ biasq,
              const float* __restrict__ biask,
              const float* __restrict__ biasv,
              bf16* __restrict__ Qo,                // [B,H,S,HD]
              bf16* __restrict__ Ko,                // [B,H,S,HD]
              bf16* __restrict__ Vt)                // [B,H,HD,S]
{
    __shared__ bf16 sA[2][128 * 32];                // 8 KB per buffer
    __shared__ bf16 sB[2][128 * 32];                // 8 KB per buffer

    const int tid  = threadIdx.x;
    const int lane = tid & 31;
    const int ln16 = lane & 15;
    const int hihf = lane >> 4;                     // 0 / 1 lane-half
    const int wid  = tid >> 5;
    const int wm   = wid & 3;                       // wave M group (4 x 32 rows)
    const int wn   = wid >> 2;                      // wave N group (2 x 64 cols)

    const int MB = NTOK / 128;                      // 64
    const int NB = En / 128;                        // 8
    const int p  = blockIdx.x / (MB * NB);          // 0=Q 1=K 2=V
    const int r  = blockIdx.x % (MB * NB);
    const int m0 = (r / NB) * 128;
    const int n0 = (r % NB) * 128;
    const bf16* wp = wb + (size_t)p * En * En;

    // per-stage copy: 128 rows x 32 k of A and of B = 2x8KB = 1024 b128 chunks;
    // 256 threads x (2 A + 2 B) async b128 each.
    const unsigned baseA0 = (unsigned)(size_t)&sA[0][0];
    const unsigned baseA1 = (unsigned)(size_t)&sA[1][0];
    const unsigned baseB0 = (unsigned)(size_t)&sB[0][0];
    const unsigned baseB1 = (unsigned)(size_t)&sB[1][0];

    #define ISSUE_STAGE(bufsel, k0)                                            \
        do {                                                                   \
            _Pragma("unroll")                                                  \
            for (int j = 0; j < 2; ++j) {                                      \
                const int c   = tid + j * 256;                                 \
                const int row = c >> 2;                                        \
                const int ko  = (c & 3) * 8;                                   \
                const unsigned off = (unsigned)(c * 16);                       \
                async_b128(((bufsel) ? baseA1 : baseA0) + off,                 \
                           xb + (size_t)(m0 + row) * En + (k0) + ko);          \
                async_b128(((bufsel) ? baseB1 : baseB0) + off,                 \
                           wp + (size_t)(n0 + row) * En + (k0) + ko);          \
            }                                                                  \
        } while (0)

    v8f acc[2][4] = {};

    ISSUE_STAGE(0, 0);
    const int NSTAGE = En / 32;                     // 32
    for (int s = 0; s < NSTAGE; ++s) {
        const int buf = s & 1;
        if (s + 1 < NSTAGE) {
            ISSUE_STAGE(buf ^ 1, (s + 1) * 32);
            asm volatile("s_wait_asynccnt 0x4" ::: "memory");   // stage s done
        } else {
            asm volatile("s_wait_asynccnt 0x0" ::: "memory");
        }
        __syncthreads();                            // stage-s data visible

        // A fragments for this wave's 32 rows
        v16bf afr[2];
#pragma unroll
        for (int mf = 0; mf < 2; ++mf) {
            const bf16* ap = &sA[buf][(wm * 32 + mf * 16 + ln16) * 32 + hihf * 8];
            afr[mf] = pack16(*(const v8bf*)ap, *(const v8bf*)(ap + 16));
        }
#pragma unroll
        for (int t = 0; t < 4; ++t) {
            const bf16* bp = &sB[buf][(wn * 64 + t * 16 + ln16) * 32 + hihf * 8];
            v16bf bfr = pack16(*(const v8bf*)bp, *(const v8bf*)(bp + 16));
#pragma unroll
            for (int mf = 0; mf < 2; ++mf)
                acc[mf][t] = wmma_bf16(afr[mf], bfr, acc[mf][t]);
        }
        __syncthreads();                            // done reading buf
    }
    #undef ISSUE_STAGE

    const float* bias = (p == 0) ? biasq : (p == 1) ? biask : biasv;
#pragma unroll
    for (int t = 0; t < 4; ++t) {
        const int col = n0 + wn * 64 + t * 16 + ln16;   // == h*64 + hd
        const float bv = bias[col];
        const int h  = col >> 6;
        const int hd = col & 63;
#pragma unroll
        for (int mf = 0; mf < 2; ++mf) {
#pragma unroll
            for (int e = 0; e < 8; ++e) {
                const int m = m0 + wm * 32 + mf * 16 + e + hihf * 8;
                const int b = m / Sn, sidx = m % Sn;
                const bf16 o = (bf16)(acc[mf][t][e] + bv);
                if (p == 0)      Qo[(((size_t)b * Hn + h) * Sn + sidx) * HDn + hd] = o;
                else if (p == 1) Ko[(((size_t)b * Hn + h) * Sn + sidx) * HDn + hd] = o;
                else             Vt[(((size_t)b * Hn + h) * HDn + hd) * Sn + sidx] = o;
            }
        }
    }
}

// ---------------------------------------------------------------------------
// phase 3: causal flash attention. One wave per (b, h, 16-query tile).
// 32 keys per step; online softmax; P goes through LDS to become an A-frag.
// ---------------------------------------------------------------------------
__global__ __launch_bounds__(32)
void flash_attn(const bf16* __restrict__ Q,         // [B,H,S,HD]
                const bf16* __restrict__ K,         // [B,H,S,HD]
                const bf16* __restrict__ Vt,        // [B,H,HD,S]
                float* __restrict__ out)            // [B,S,E]
{
    __shared__ bf16 ptile[16 * 32];                 // P tile, row-major

    const int lane = threadIdx.x & 31;
    const int ln16 = lane & 15;
    const int hihf = lane >> 4;
    const int blk  = blockIdx.x;
    const int bh   = blk >> 7;                      // / (S/16)
    const int qt   = blk & 127;
    const int q0   = qt * 16;
    const int b    = bh >> 4;
    const int h    = bh & 15;

    const bf16* Qb = Q  + (size_t)bh * Sn * HDn;
    const bf16* Kb = K  + (size_t)bh * Sn * HDn;
    const bf16* Vb = Vt + (size_t)bh * HDn * Sn;

    // Q fragments (16x64 = two 16x32 A-frags), resident for whole loop
    v16bf aq[2];
#pragma unroll
    for (int c = 0; c < 2; ++c) {
        const bf16* qp = Qb + (size_t)(q0 + ln16) * HDn + c * 32 + hihf * 8;
        aq[c] = pack16(*(const v8bf*)qp, *(const v8bf*)(qp + 16));
    }

    float mrow[8], lrow[8];
    v8f oacc[4] = {};
#pragma unroll
    for (int e = 0; e < 8; ++e) { mrow[e] = NEG_INF; lrow[e] = 0.0f; }

    const float inv_scale = 0.125f;                 // 1/sqrt(64)
    const int ktmax = (q0 + 15) / 32;

    for (int kt = 0; kt <= ktmax; ++kt) {
        const int kv0 = kt * 32;

        // ---- scores S = Q * K^T  (two 16x16 C tiles, K-dim = HD = 2x32) ----
        v8f sc[2];
#pragma unroll
        for (int s = 0; s < 2; ++s) {
            const int key = kv0 + s * 16 + ln16;    // B column = key index
            const bf16* kp0 = Kb + (size_t)key * HDn + hihf * 8;
            const bf16* kp1 = kp0 + 32;
            v16bf kb0 = pack16(*(const v8bf*)kp0, *(const v8bf*)(kp0 + 16));
            v16bf kb1 = pack16(*(const v8bf*)kp1, *(const v8bf*)(kp1 + 16));
            v8f c = {};
            c = wmma_bf16(aq[0], kb0, c);
            c = wmma_bf16(aq[1], kb1, c);
            sc[s] = c;
        }

        // ---- scale, causal mask, online softmax ----
        const int key0 = kv0 + ln16;
        const int key1 = kv0 + 16 + ln16;
#pragma unroll
        for (int e = 0; e < 8; ++e) {
            const int qrow = q0 + e + hihf * 8;
            float s0 = sc[0][e] * inv_scale; if (key0 > qrow) s0 = NEG_INF;
            float s1 = sc[1][e] * inv_scale; if (key1 > qrow) s1 = NEG_INF;

            float mx = fmaxf(s0, s1);
#pragma unroll
            for (int msk = 1; msk < 16; msk <<= 1)
                mx = fmaxf(mx, __shfl_xor(mx, msk, 32));
            const float mnew = fmaxf(mrow[e], mx);
            const float fac  = __expf(mrow[e] - mnew);
            mrow[e] = mnew;

            const float p0 = __expf(s0 - mnew);
            const float p1 = __expf(s1 - mnew);
            float ss = p0 + p1;
#pragma unroll
            for (int msk = 1; msk < 16; msk <<= 1)
                ss += __shfl_xor(ss, msk, 32);
            lrow[e] = lrow[e] * fac + ss;
#pragma unroll
            for (int t = 0; t < 4; ++t) oacc[t][e] *= fac;

            const int row = e + hihf * 8;           // C-layout row
            ptile[row * 32 + ln16]      = (bf16)p0;
            ptile[row * 32 + 16 + ln16] = (bf16)p1;
        }

        // ---- P (via LDS) as A-fragment, then O += P * V ----
        const bf16* pp = &ptile[ln16 * 32 + hihf * 8];
        v16bf pa = pack16(*(const v8bf*)pp, *(const v8bf*)(pp + 16));
#pragma unroll
        for (int t = 0; t < 4; ++t) {
            const int hd = t * 16 + ln16;           // B column = HD index
            const bf16* vp = Vb + (size_t)hd * Sn + kv0 + hihf * 8;
            v16bf vbf = pack16(*(const v8bf*)vp, *(const v8bf*)(vp + 16));
            oacc[t] = wmma_bf16(pa, vbf, oacc[t]);
        }
    }

    // ---- normalize and store fp32 output [B,S,H*HD] ----
#pragma unroll
    for (int t = 0; t < 4; ++t) {
        const int col = h * 64 + t * 16 + ln16;
#pragma unroll
        for (int e = 0; e < 8; ++e) {
            const int qrow = q0 + e + hihf * 8;
            out[((size_t)b * Sn + qrow) * En + col] = oacc[t][e] / lrow[e];
        }
    }
}

// ---------------------------------------------------------------------------
// launch
// ---------------------------------------------------------------------------
extern "C" void kernel_launch(void* const* d_in, const int* in_sizes, int n_in,
                              void* d_out, int out_size, void* d_ws, size_t ws_size,
                              hipStream_t stream) {
    const float* x  = (const float*)d_in[0];
    const float* Wq = (const float*)d_in[1];
    const float* bq = (const float*)d_in[2];
    const float* Wk = (const float*)d_in[3];
    const float* bk = (const float*)d_in[4];
    const float* Wv = (const float*)d_in[5];
    const float* bv = (const float*)d_in[6];
    float* out = (float*)d_out;

    // workspace layout (bf16 buffers)
    char* ws = (char*)d_ws;
    const size_t SZ_X = (size_t)NTOK * En * 2;                 // 16 MB
    const size_t SZ_W = (size_t)En * En * 2;                   // 2 MB each
    const size_t SZ_P = (size_t)Bn * Hn * Sn * HDn * 2;        // 16 MB each
    bf16* xb  = (bf16*)(ws);
    bf16* wb  = (bf16*)(ws + SZ_X);                            // [3][E][E]
    bf16* Qb  = (bf16*)(ws + SZ_X + 3 * SZ_W);
    bf16* Kb  = (bf16*)(ws + SZ_X + 3 * SZ_W + SZ_P);
    bf16* Vtb = (bf16*)(ws + SZ_X + 3 * SZ_W + 2 * SZ_P);

    // phase 1: conversions
    cvt_f32_bf16<<<1024, 256, 0, stream>>>(x,  xb,               NTOK * En);
    cvt_f32_bf16<<<512,  256, 0, stream>>>(Wq, wb,               En * En);
    cvt_f32_bf16<<<512,  256, 0, stream>>>(Wk, wb + En * En,     En * En);
    cvt_f32_bf16<<<512,  256, 0, stream>>>(Wv, wb + 2 * En * En, En * En);

    // phase 2: QKV projection — 3 * 64 * 8 blocks of 128x128, 256 threads
    qkv_gemm<<<3 * (NTOK / 128) * (En / 128), 256, 0, stream>>>(
        xb, wb, bq, bk, bv, Qb, Kb, Vtb);

    // phase 3: attention — B*H*(S/16) waves, 1 wave per block
    flash_attn<<<Bn * Hn * (Sn / 16), 32, 0, stream>>>(Qb, Kb, Vtb, out);
}
